// RMVSNet_11622181503521
// MI455X (gfx1250) — matrix operand order repaired and checked by reading, not compiled, not run on hardware
//
#include <hip/hip_runtime.h>
#include <hip/hip_bf16.h>

// RMVSNet forward for MI455X (gfx1250).
// All conv / deconv / GRU-conv layers run as implicit-GEMM on
// v_wmma_f32_16x16x32_f16 (f16 in, f32 accumulate). Weight (A) tiles are
// staged into LDS by the Tensor Data Mover (tensor_load_to_lds +
// s_wait_tensorcnt) when the tile is fully interior; im2col (B) tiles are
// gathered manually. One wave per block so EXEC is all-ones at the WMMA;
// each wave register-blocks an MTxNT grid of 16x16 tiles (4 WMMAs per
// K-chunk for the big layers).

typedef __attribute__((ext_vector_type(16))) _Float16 v16h;
typedef __attribute__((ext_vector_type(8)))  _Float16 v8h;
typedef __attribute__((ext_vector_type(8)))  float    v8f;
typedef __attribute__((ext_vector_type(4)))  float    v4f;
typedef __attribute__((ext_vector_type(4)))  unsigned int v4u;
typedef __attribute__((ext_vector_type(8)))  int      v8i;
typedef __attribute__((ext_vector_type(4)))  int      v4i;

#if defined(__has_builtin)
#if __has_builtin(__builtin_amdgcn_tensor_load_to_lds) && \
    __has_builtin(__builtin_amdgcn_s_wait_tensorcnt)
#define USE_TDM 1
#endif
#endif
#ifndef USE_TDM
#define USE_TDM 0
#endif

#define S0 (320*448)
#define S1 (160*224)
#define S2 (80*112)
#define S3 (40*56)
#define S4 (20*28)
#define DNUM 64

// ---------------------------------------------------------------------------
// Implicit-GEMM conv / deconv kernel (register-blocked WMMA + TDM A-stage).
//   out[n][co][oy][ox] = bias[co] + sum_{ci,ky,kx} in[n][ci][iy][ix] * w[co][ci][ky][kx]
// conv  (deconv=0): iy = oy*stride + ky - pad
// deconv(deconv=1): iy = oy - pad + ky, valid only if even; iy/=2  (lhs-dilated
//                   conv_transpose, stride 2, SAME, k=3 -> pad=2)
// Input may be a channel-concat of two tensors (C1 from in1, C2 from in2).
// GEMM: M=Cout(tile 16*MT) x N=N*Hout*Wout(tile 16*NT), K=Cin*Kk*Kk (chunks of 32).
// Weight matrix is [Cout][Ktot] contiguous, so the A tile is a plain 2D tile:
// tile_dim0=32 (k), tile_dim1=16*MT (rows), row stride Ktot -> TDM descriptor.
// ---------------------------------------------------------------------------
template <int MT, int NT>
__global__ void __launch_bounds__(32) conv_wmma_kernel(
    const float* __restrict__ in1, int C1,
    const float* __restrict__ in2, int C2,
    const float* __restrict__ wgt, const float* __restrict__ bias,
    float* __restrict__ out,
    int N, int H, int W, int Cout, int Kk, int stride, int pad,
    int Hout, int Wout, int deconv)
{
  // Single static LDS block so the TDM lds_addr byte offset is known:
  //   [0, NT*1024)                : lB, f16 [p][k]   (B^T: pixel x K)
  //   [NT*1024, NT*1024+MT*2048)  : lAf, f32 [m][k]  (A: Cout x K)
  __shared__ __align__(16) unsigned char smem[NT*16*32*2 + MT*16*32*4];
  _Float16* lB  = (_Float16*)smem;
  float*    lAf = (float*)(smem + NT*16*32*2);
  const unsigned LDS_A_OFF = (unsigned)(NT*16*32*2);

  const int lane = threadIdx.x;
  const int Cin  = C1 + C2;
  const int KK   = Kk*Kk;
  const int Ktot = Cin*KK;
  const int HWo  = Hout*Wout;
  const int HW   = H*W;
  const int pTile  = blockIdx.x * (16*NT);
  const int coTile = blockIdx.y * (16*MT);

  v8f acc[MT][NT] = {};

  for (int k0 = 0; k0 < Ktot; k0 += 32) {
    __syncthreads();                 // protect LDS from previous iter's reads
    // ---- stage A (weights) as f32: lAf[m*32+kk] ----
#if USE_TDM
    const bool fullTile = (k0 + 32 <= Ktot) && (coTile + MT*16 <= Cout);
    if (fullTile) {
      // 2D TDM descriptor (D#), groups per CDNA5 ISA.
      unsigned long long ga =
          (unsigned long long)(uintptr_t)(wgt + (size_t)coTile*(size_t)Ktot + k0);
      v4u g0 = {
        1u,                                            // count=1, user D#
        LDS_A_OFF,                                     // lds_addr (bytes)
        (unsigned)ga,                                  // global_addr[31:0]
        (unsigned)((ga >> 32) & 0x01FFFFFFu) | (2u<<30) // addr[56:32] | type=2
      };
      v8i g1 = {
        (int)(2u << 16),                               // data_size=4B, no flags
        (int)(((unsigned)Ktot & 0xFFFFu) << 16),       // tensor_dim0[15:0]
        (int)(((unsigned)Ktot >> 16) |
              (((unsigned)Cout & 0xFFFFu) << 16)),     // dim0 hi | tensor_dim1 lo
        (int)(((unsigned)Cout >> 16) | (32u << 16)),   // dim1 hi | tile_dim0=32
        (int)(MT*16),                                  // tile_dim1 | tile_dim2=0
        (int)Ktot,                                     // tensor_dim0_stride lo
        0,                                             // stride hi | dim1_stride lo
        0
      };
      v4i g2 = {0,0,0,0}, g3 = {0,0,0,0};              // 2D tensor: unused
      v8i g4 = {0,0,0,0,0,0,0,0};                      // unused high-dim group
      __builtin_amdgcn_tensor_load_to_lds(g0, g1, g2, g3, g4, 0);
    } else
#endif
    {
      #pragma unroll 1
      for (int e = lane; e < MT*512; e += 32) {
        int m  = e >> 5;
        int kk = e & 31;
        int co = coTile + m;
        int kg = k0 + kk;
        float v = 0.f;
        if (co < Cout && kg < Ktot) {
          int ci = kg / KK;
          int r  = kg - ci*KK;
          v = wgt[(co*Cin + ci)*KK + r];
        }
        lAf[e] = v;
      }
    }
    // prefetch next weight chunk toward L2 (global_prefetch_b8)
    if (k0 + 32 < Ktot)
      __builtin_prefetch(wgt + (size_t)coTile*Cin*KK + (k0 + 32), 0, 1);
    // ---- stage B (im2col gather): lB[p*32+kk], f16 ----
    #pragma unroll 1
    for (int e = lane; e < NT*512; e += 32) {
      int p  = e >> 5;
      int kk = e & 31;
      int kg = k0 + kk;
      int pg = pTile + p;
      float v = 0.f;
      if (kg < Ktot && pg < N*HWo) {
        int nb = pg / HWo;  int pr = pg - nb*HWo;
        int oy = pr / Wout; int ox = pr - oy*Wout;
        int ci = kg / KK;   int r  = kg - ci*KK;
        int ky = r / Kk;    int kx = r - ky*Kk;
        int iy, ix; bool ok;
        if (deconv) {
          iy = oy - pad + ky; ix = ox - pad + kx;
          ok = ((iy | ix) >= 0) && ((iy & 1) == 0) && ((ix & 1) == 0);
          iy >>= 1; ix >>= 1;
          ok = ok && (iy < H) && (ix < W);
        } else {
          iy = oy*stride + ky - pad; ix = ox*stride + kx - pad;
          ok = (iy >= 0) && (iy < H) && (ix >= 0) && (ix < W);
        }
        if (ok) {
          if (ci < C1) v = in1[(nb*C1 + ci)*HW + iy*W + ix];
          else         v = in2[(nb*C2 + (ci - C1))*HW + iy*W + ix];
        }
      }
      lB[e] = (_Float16)v;
    }
    __syncthreads();                 // ds stores visible
#if USE_TDM
    __builtin_amdgcn_s_wait_tensorcnt(0);  // TDM tile landed in LDS
#endif
    // ---- pack fragments per CDNA5 WMMA layout ----
    // A 16x32 f16: lane row m=lane&15; lo-half lanes: K 0-7 (v0-3) & 16-23
    //              (v4-7); hi-half lanes: K 8-15 & 24-31.
    // B 32x16 f16: lane column n=lane&15; lo-half: K 0-7 & 16-23; hi-half: K 8-15 & 24-31.
    const int mrow = lane & 15;
    const int hi   = lane >> 4;
    v16h afr[MT], bfr[NT];
    #pragma unroll
    for (int mt = 0; mt < MT; ++mt) {
      const float* row = &lAf[(mt*16 + mrow)*32];
      v4f c0 = *(const v4f*)&row[hi*8];
      v4f c1 = *(const v4f*)&row[hi*8 + 4];
      v4f c2 = *(const v4f*)&row[16 + hi*8];
      v4f c3 = *(const v4f*)&row[16 + hi*8 + 4];
      #pragma unroll
      for (int i = 0; i < 4; ++i) {
        afr[mt][i]    = (_Float16)c0[i];
        afr[mt][4+i]  = (_Float16)c1[i];
        afr[mt][8+i]  = (_Float16)c2[i];
        afr[mt][12+i] = (_Float16)c3[i];
      }
    }
    #pragma unroll
    for (int nt = 0; nt < NT; ++nt) {
      const _Float16* row = &lB[(nt*16 + mrow)*32];
      v8h b0 = *(const v8h*)&row[hi*8];
      v8h b1 = *(const v8h*)&row[16 + hi*8];
      #pragma unroll
      for (int i = 0; i < 8; ++i) { bfr[nt][i] = b0[i]; bfr[nt][8+i] = b1[i]; }
    }
    #pragma unroll
    for (int mt = 0; mt < MT; ++mt)
      #pragma unroll
      for (int nt = 0; nt < NT; ++nt)
        acc[mt][nt] = __builtin_amdgcn_wmma_f32_16x16x32_f16(
            /*neg_a=*/false, afr[mt], /*neg_b=*/false, bfr[nt],
            /*c_mod=*/(short)0, acc[mt][nt],
            /*reuse_a=*/false, /*reuse_b=*/false);
  }

  // ---- write D: lane holds column p=lane&15; VGPR r holds row r+8*hi ----
  #pragma unroll
  for (int nt = 0; nt < NT; ++nt) {
    int p = pTile + nt*16 + (lane & 15);
    if (p < N*HWo) {
      int nb = p / HWo; int pr = p - nb*HWo;
      #pragma unroll
      for (int mt = 0; mt < MT; ++mt) {
        #pragma unroll
        for (int r = 0; r < 8; ++r) {
          int m  = mt*16 + r + ((lane >> 4) << 3);
          int co = coTile + m;
          if (co < Cout)
            out[(nb*Cout + co)*HWo + pr] = acc[mt][nt][r] + bias[co];
        }
      }
    }
  }
}

// ---------------------------------------------------------------------------
// GroupNorm (+optional ReLU), in-place. One block per (n, group).
// ---------------------------------------------------------------------------
__global__ void __launch_bounds__(256) gn_relu_kernel(
    float* __restrict__ y, const float* __restrict__ gam,
    const float* __restrict__ bet, int C, int HW, int groups, int relu)
{
  const int bid = blockIdx.x;
  const int grp = bid % groups;
  const int n   = bid / groups;
  const int cpg = C / groups;
  const long long base = ((long long)n*C + (long long)grp*cpg) * HW;
  const int len = cpg * HW;
  __shared__ float ss[256], sq[256];
  float s = 0.f, s2 = 0.f;
  for (int i = threadIdx.x; i < len; i += 256) {
    float v = y[base + i]; s += v; s2 += v*v;
  }
  ss[threadIdx.x] = s; sq[threadIdx.x] = s2;
  __syncthreads();
  for (int w = 128; w > 0; w >>= 1) {
    if (threadIdx.x < w) {
      ss[threadIdx.x] += ss[threadIdx.x + w];
      sq[threadIdx.x] += sq[threadIdx.x + w];
    }
    __syncthreads();
  }
  const float inv_len = 1.f / (float)len;
  const float mu  = ss[0] * inv_len;
  const float var = sq[0] * inv_len - mu*mu;
  const float rs  = rsqrtf(var + 1e-5f);
  for (int i = threadIdx.x; i < len; i += 256) {
    int c = grp*cpg + i / HW;
    float v = (y[base + i] - mu) * rs * gam[c] + bet[c];
    if (relu) v = fmaxf(v, 0.f);
    y[base + i] = v;
  }
}

// ---------------------------------------------------------------------------
// Homographies: Hs[v][d] = Kf[v] * (Rrel + trel*n^T/depth_d) * inv(Kf[0])
// ---------------------------------------------------------------------------
__device__ __forceinline__ void mat3mul(const float* A, const float* B, float* Cm) {
  #pragma unroll
  for (int i = 0; i < 3; ++i)
    #pragma unroll
    for (int j = 0; j < 3; ++j)
      Cm[i*3+j] = A[i*3+0]*B[0*3+j] + A[i*3+1]*B[1*3+j] + A[i*3+2]*B[2*3+j];
}
__device__ __forceinline__ void inv3(const float* m, float* o) {
  float a=m[0],b=m[1],c=m[2],d=m[3],e=m[4],f=m[5],g=m[6],h=m[7],i=m[8];
  float A =  e*i - f*h;
  float B = -(d*i - f*g);
  float Cc=  d*h - e*g;
  float det = a*A + b*B + c*Cc;
  float r = 1.f / det;
  o[0]=A*r;             o[1]=-(b*i-c*h)*r;  o[2]=(b*f-c*e)*r;
  o[3]=B*r;             o[4]=(a*i-c*g)*r;   o[5]=-(a*f-c*d)*r;
  o[6]=Cc*r;            o[7]=-(a*h-b*g)*r;  o[8]=(a*e-b*d)*r;
}

__global__ void __launch_bounds__(256) homography_kernel(
    const float* __restrict__ Kin, const float* __restrict__ Ein,
    const float* __restrict__ dstart, const float* __restrict__ dint,
    float* __restrict__ Hs)
{
  int t = threadIdx.x;
  if (t >= 3*DNUM) return;
  int v = t / DNUM;
  int d = t - v*DNUM;
  float R[9], R0[9], tv[3], t0[3];
  #pragma unroll
  for (int i = 0; i < 3; ++i) {
    #pragma unroll
    for (int j = 0; j < 3; ++j) {
      R[i*3+j]  = Ein[v*16 + i*4 + j];
      R0[i*3+j] = Ein[i*4 + j];
    }
    tv[i] = Ein[v*16 + i*4 + 3];
    t0[i] = Ein[i*4 + 3];
  }
  float Rr[9];                     // Rrel = R * R0^T
  #pragma unroll
  for (int i = 0; i < 3; ++i)
    #pragma unroll
    for (int j = 0; j < 3; ++j)
      Rr[i*3+j] = R[i*3+0]*R0[j*3+0] + R[i*3+1]*R0[j*3+1] + R[i*3+2]*R0[j*3+2];
  float tr[3];
  #pragma unroll
  for (int i = 0; i < 3; ++i)
    tr[i] = tv[i] - (Rr[i*3+0]*t0[0] + Rr[i*3+1]*t0[1] + Rr[i*3+2]*t0[2]);
  const float depth = dstart[0] + (float)d * dint[0];
  float M[9];
  #pragma unroll
  for (int i = 0; i < 9; ++i) M[i] = Rr[i];
  #pragma unroll
  for (int i = 0; i < 3; ++i) M[i*3+2] += tr[i] / depth;
  float Kf[9], Kf0[9], Kf0i[9], T[9], Hh[9];
  #pragma unroll
  for (int i = 0; i < 3; ++i)
    #pragma unroll
    for (int j = 0; j < 3; ++j) {
      float sc = (i < 2) ? 0.25f : 1.f;
      Kf[i*3+j]  = Kin[v*9 + i*3 + j] * sc;
      Kf0[i*3+j] = Kin[i*3 + j] * sc;
    }
  inv3(Kf0, Kf0i);
  mat3mul(Kf, M, T);
  mat3mul(T, Kf0i, Hh);
  #pragma unroll
  for (int i = 0; i < 9; ++i) Hs[(v*DNUM + d)*9 + i] = Hh[i];
}

// ---------------------------------------------------------------------------
// Plane-sweep warp + variance cost (negated for the GRU input).
// ncost[d][c][y][x] = mean_v(w)^2 - mean_v(w^2)     (== -(var))
// ---------------------------------------------------------------------------
__device__ __forceinline__ float tap(const float* img, int y, int x, int h, int w) {
  if (x < 0 || x >= w || y < 0 || y >= h) return 0.f;
  return img[y*w + x];
}

__global__ void __launch_bounds__(256) warpcost_kernel(
    const float* __restrict__ f, const float* __restrict__ Hs,
    float* __restrict__ ncost, int C, int h, int w)
{
  int idx = blockIdx.x*256 + threadIdx.x;
  int total = DNUM * h * w;
  if (idx >= total) return;
  int d = idx / (h*w);
  int pr = idx - d*(h*w);
  int y = pr / w;
  int x = pr - y*w;
  float wwx[3], wwy[3];
  int x0[3], y0[3];
  #pragma unroll
  for (int v = 0; v < 3; ++v) {
    const float* Hm = &Hs[(v*DNUM + d)*9];
    float px = Hm[0]*(float)x + Hm[1]*(float)y + Hm[2];
    float py = Hm[3]*(float)x + Hm[4]*(float)y + Hm[5];
    float pz = Hm[6]*(float)x + Hm[7]*(float)y + Hm[8] + 1e-6f;
    float xv = px / pz, yv = py / pz;
    float xf = floorf(xv), yf = floorf(yv);
    wwx[v] = xv - xf; wwy[v] = yv - yf;
    x0[v] = (int)xf;  y0[v] = (int)yf;
  }
  for (int c = 0; c < C; ++c) {
    float s = 0.f, s2 = 0.f;
    #pragma unroll
    for (int v = 0; v < 3; ++v) {
      const float* img = f + (size_t)(v*C + c) * (h*w);
      float v00 = tap(img, y0[v],   x0[v],   h, w);
      float v01 = tap(img, y0[v],   x0[v]+1, h, w);
      float v10 = tap(img, y0[v]+1, x0[v],   h, w);
      float v11 = tap(img, y0[v]+1, x0[v]+1, h, w);
      float a = wwx[v], b = wwy[v];
      float val = v00*(1.f-a)*(1.f-b) + v01*a*(1.f-b)
                + v10*(1.f-a)*b       + v11*a*b;
      s += val; s2 += val*val;
    }
    float av  = s  * (1.f/3.f);
    float av2 = s2 * (1.f/3.f);
    ncost[((size_t)d*C + c)*(h*w) + pr] = av*av - av2;
  }
}

// ---------------------------------------------------------------------------
// Elementwise helpers
// ---------------------------------------------------------------------------
__global__ void ew_sigmoid_kernel(float* x, int n) {
  int i = blockIdx.x*256 + threadIdx.x;
  if (i < n) x[i] = 1.f / (1.f + __expf(-x[i]));
}
__global__ void ew_tanh_kernel(float* x, int n) {
  int i = blockIdx.x*256 + threadIdx.x;
  if (i < n) x[i] = tanhf(x[i]);
}
__global__ void ew_mul_kernel(const float* a, const float* b, float* o, int n) {
  int i = blockIdx.x*256 + threadIdx.x;
  if (i < n) o[i] = a[i] * b[i];
}
__global__ void ew_gru_kernel(const float* u, const float* c, float* h, int n) {
  int i = blockIdx.x*256 + threadIdx.x;
  if (i < n) h[i] = u[i]*h[i] + (1.f - u[i])*c[i];
}
__global__ void ew_zero_kernel(float* x, int n) {
  int i = blockIdx.x*256 + threadIdx.x;
  if (i < n) x[i] = 0.f;
}

// ---------------------------------------------------------------------------
// Softmax over depth for each pixel. out shape (1, D, 1, h, w).
// ---------------------------------------------------------------------------
__global__ void __launch_bounds__(256) softmax_kernel(
    const float* __restrict__ reg, float* __restrict__ out, int HW)
{
  int p = blockIdx.x*256 + threadIdx.x;
  if (p >= HW) return;
  float mx = -3.0e38f;
  for (int d = 0; d < DNUM; ++d) mx = fmaxf(mx, reg[d*HW + p]);
  float s = 0.f;
  for (int d = 0; d < DNUM; ++d) s += __expf(reg[d*HW + p] - mx);
  float inv = 1.f / s;
  for (int d = 0; d < DNUM; ++d) out[d*HW + p] = __expf(reg[d*HW + p] - mx) * inv;
}

// ---------------------------------------------------------------------------
// Host-side orchestration
// ---------------------------------------------------------------------------
static inline void conv_launch(hipStream_t st, const float* in1, int C1,
    const float* in2, int C2, const float* w, const float* b, float* out,
    int N, int H, int W, int Cout, int K, int stride, int pad,
    int Hout, int Wout, int deconv)
{
  int npix = N*Hout*Wout;
  if (Cout >= 32) {
    dim3 g((npix + 31)/32, (Cout + 31)/32, 1);
    conv_wmma_kernel<2,2><<<g, dim3(32,1,1), 0, st>>>(
        in1, C1, in2 ? in2 : in1, C2, w, b, out,
        N, H, W, Cout, K, stride, pad, Hout, Wout, deconv);
  } else {
    dim3 g((npix + 31)/32, (Cout + 15)/16, 1);
    conv_wmma_kernel<1,2><<<g, dim3(32,1,1), 0, st>>>(
        in1, C1, in2 ? in2 : in1, C2, w, b, out,
        N, H, W, Cout, K, stride, pad, Hout, Wout, deconv);
  }
}

static inline void gn_launch(hipStream_t st, float* y, const float* g,
                             const float* be, int N, int C, int HW)
{
  int groups = C / 8; if (groups < 1) groups = 1;
  gn_relu_kernel<<<dim3(N*groups), dim3(256), 0, st>>>(y, g, be, C, HW, groups, 1);
}

static inline void ew_grid(int n, dim3& g) { g = dim3((n + 255)/256, 1, 1); }

extern "C" void kernel_launch(void* const* d_in, const int* in_sizes, int n_in,
                              void* d_out, int out_size, void* d_ws, size_t ws_size,
                              hipStream_t stream)
{
  (void)in_sizes; (void)n_in; (void)out_size; (void)ws_size;
  const float* images = (const float*)d_in[0];
  const float* Kmat   = (const float*)d_in[1];
  const float* Emat   = (const float*)d_in[2];
  const float* dstart = (const float*)d_in[3];
  const float* dint   = (const float*)d_in[4];
  // d_in[5] = depth_num (int, == 64, compiled in as DNUM)

  // params flattened jax-tree style (sorted dict keys), appended after the 6
  // top-level inputs:
  //  6..11  gru1: bc,br,bu,wc,wr,wu     12..17 gru2    18..23 gru3
  //  24 prob.b  25 prob.w
  //  26.. unet convs in sorted-name order, leaves (b,be,g,w); c10_2 has (b,w).
  auto P = [&](int i) { return (const float*)d_in[i]; };
  const int C0_1=26, C10_0=30, C10_1=34, C10_2b=38, C10_2w=39,
            C1_0=40, C1_1=44, C2_0=48, C2_1=52, C3_0=56, C3_1=60,
            C4_0=64, C4_1=68, C5_0=72, C5_1=76, C6_0=80, C6_1=84,
            C7_0=88, C7_1=92, C8_0=96, C8_1=100, C9_0=104, C9_1=108, C9_2=112;

  // bump-allocate workspace (floats)
  float* base = (float*)d_ws;
  size_t off = 0;
  auto alloc = [&](size_t n) { float* p = base + off; off += n; return p; };
  float* c01  = alloc((size_t)3*8*S0);
  float* c10b = alloc((size_t)3*16*S1);
  float* c20b = alloc((size_t)3*32*S2);
  float* c30b = alloc((size_t)3*64*S3);
  float* c40b = alloc((size_t)3*128*S4);
  float* c11  = alloc((size_t)3*16*S1);
  float* c21  = alloc((size_t)3*32*S2);
  float* c31  = alloc((size_t)3*64*S3);
  float* c41  = alloc((size_t)3*128*S4);
  float* d5   = alloc((size_t)3*64*S3);
  float* u5   = alloc((size_t)3*64*S3);
  float* d6   = alloc((size_t)3*32*S2);
  float* u6   = alloc((size_t)3*32*S2);
  float* d7   = alloc((size_t)3*16*S1);
  float* u7   = alloc((size_t)3*16*S1);
  float* d8   = alloc((size_t)3*8*S0);
  float* u8   = alloc((size_t)3*8*S0);
  float* t9a  = alloc((size_t)3*16*S1);
  float* t9b  = alloc((size_t)3*16*S1);
  float* t9c  = alloc((size_t)3*16*S1);
  float* t10a = alloc((size_t)3*32*S2);
  float* t10b = alloc((size_t)3*32*S2);
  float* fbuf = alloc((size_t)3*32*S2);
  float* HsB  = alloc((size_t)3*DNUM*9);
  float* ncost= alloc((size_t)DNUM*32*S2);
  float* h1   = alloc((size_t)16*S2);
  float* h2   = alloc((size_t)4*S2);
  float* h3   = alloc((size_t)2*S2);
  float* tA   = alloc((size_t)16*S2);
  float* tB   = alloc((size_t)16*S2);
  float* tC   = alloc((size_t)16*S2);
  float* tD   = alloc((size_t)16*S2);
  float* regb = alloc((size_t)DNUM*S2);

#define CGR(idx, in1p, c1, in2p, c2, outp, N_, H_, W_, CO, K_, S_, PAD, HO, WO, DEC) \
  do {                                                                               \
    conv_launch(stream, in1p, c1, in2p, c2, P((idx)+3), P((idx)+0), outp,            \
                N_, H_, W_, CO, K_, S_, PAD, HO, WO, DEC);                           \
    gn_launch(stream, outp, P((idx)+2), P((idx)+1), N_, CO, (HO)*(WO));              \
  } while (0)

  // ---------------- UNet ----------------
  CGR(C0_1, images,3, nullptr,0, c01,  3,320,448,   8,3,1,1, 320,448, 0);
  CGR(C1_0, images,3, nullptr,0, c10b, 3,320,448,  16,3,2,0, 160,224, 0);
  CGR(C2_0, c10b,16,  nullptr,0, c20b, 3,160,224,  32,3,2,0,  80,112, 0);
  CGR(C3_0, c20b,32,  nullptr,0, c30b, 3, 80,112,  64,3,2,0,  40, 56, 0);
  CGR(C4_0, c30b,64,  nullptr,0, c40b, 3, 40, 56, 128,3,2,0,  20, 28, 0);
  CGR(C1_1, c10b,16,  nullptr,0, c11,  3,160,224,  16,3,1,1, 160,224, 0);
  CGR(C2_1, c20b,32,  nullptr,0, c21,  3, 80,112,  32,3,1,1,  80,112, 0);
  CGR(C3_1, c30b,64,  nullptr,0, c31,  3, 40, 56,  64,3,1,1,  40, 56, 0);
  CGR(C4_1, c40b,128, nullptr,0, c41,  3, 20, 28, 128,3,1,1,  20, 28, 0);
  CGR(C5_0, c41,128,  nullptr,0, d5,   3, 20, 28,  64,3,2,2,  40, 56, 1);
  CGR(C5_1, d5,64,    c31,64,    u5,   3, 40, 56,  64,3,1,1,  40, 56, 0);
  CGR(C6_0, u5,64,    nullptr,0, d6,   3, 40, 56,  32,3,2,2,  80,112, 1);
  CGR(C6_1, d6,32,    c21,32,    u6,   3, 80,112,  32,3,1,1,  80,112, 0);
  CGR(C7_0, u6,32,    nullptr,0, d7,   3, 80,112,  16,3,2,2, 160,224, 1);
  CGR(C7_1, d7,16,    c11,16,    u7,   3,160,224,  16,3,1,1, 160,224, 0);
  CGR(C8_0, u7,16,    nullptr,0, d8,   3,160,224,   8,3,2,2, 320,448, 1);
  CGR(C8_1, d8,8,     c01,8,     u8,   3,320,448,   8,3,1,1, 320,448, 0);
  CGR(C9_0, u8,8,     nullptr,0, t9a,  3,320,448,  16,5,2,1, 160,224, 0);
  CGR(C9_1, t9a,16,   nullptr,0, t9b,  3,160,224,  16,3,1,1, 160,224, 0);
  CGR(C9_2, t9b,16,   nullptr,0, t9c,  3,160,224,  16,3,1,1, 160,224, 0);
  CGR(C10_0,t9c,16,   nullptr,0, t10a, 3,160,224,  32,5,2,1,  80,112, 0);
  CGR(C10_1,t10a,32,  nullptr,0, t10b, 3, 80,112,  32,3,1,1,  80,112, 0);
  conv_launch(stream, t10b,32, nullptr,0, P(C10_2w), P(C10_2b), fbuf,
              3, 80,112, 32,3,1,1, 80,112, 0);   // no GN / ReLU

  // ---------------- homographies + cost volume ----------------
  homography_kernel<<<dim3(1), dim3(256), 0, stream>>>(Kmat, Emat, dstart, dint, HsB);
  {
    int total = DNUM * S2;
    warpcost_kernel<<<dim3((total+255)/256), dim3(256), 0, stream>>>(
        fbuf, HsB, ncost, 32, 80, 112);
  }

  // ---------------- GRU regularization scan ----------------
  dim3 g16; ew_grid(16*S2, g16);
  dim3 g4;  ew_grid(4*S2,  g4);
  dim3 g2;  ew_grid(2*S2,  g2);
  ew_zero_kernel<<<g16, 256, 0, stream>>>(h1, 16*S2);
  ew_zero_kernel<<<g4,  256, 0, stream>>>(h2, 4*S2);
  ew_zero_kernel<<<g2,  256, 0, stream>>>(h3, 2*S2);

  for (int d = 0; d < DNUM; ++d) {
    const float* xd = ncost + (size_t)d * 32 * S2;   // -cost[d], 32ch 80x112
    // GRU1: x=xd(32), h=h1(16)
    conv_launch(stream, xd,32, h1,16, P(10), P(7), tA, 1,80,112, 16,3,1,1, 80,112, 0);
    ew_sigmoid_kernel<<<g16,256,0,stream>>>(tA, 16*S2);
    conv_launch(stream, xd,32, h1,16, P(11), P(8), tB, 1,80,112, 16,3,1,1, 80,112, 0);
    ew_sigmoid_kernel<<<g16,256,0,stream>>>(tB, 16*S2);
    ew_mul_kernel<<<g16,256,0,stream>>>(tA, h1, tC, 16*S2);
    conv_launch(stream, xd,32, tC,16, P(9), P(6), tD, 1,80,112, 16,3,1,1, 80,112, 0);
    ew_tanh_kernel<<<g16,256,0,stream>>>(tD, 16*S2);
    ew_gru_kernel<<<g16,256,0,stream>>>(tB, tD, h1, 16*S2);
    // GRU2: x=h1(16), h=h2(4)
    conv_launch(stream, h1,16, h2,4, P(16), P(13), tA, 1,80,112, 4,3,1,1, 80,112, 0);
    ew_sigmoid_kernel<<<g4,256,0,stream>>>(tA, 4*S2);
    conv_launch(stream, h1,16, h2,4, P(17), P(14), tB, 1,80,112, 4,3,1,1, 80,112, 0);
    ew_sigmoid_kernel<<<g4,256,0,stream>>>(tB, 4*S2);
    ew_mul_kernel<<<g4,256,0,stream>>>(tA, h2, tC, 4*S2);
    conv_launch(stream, h1,16, tC,4, P(15), P(12), tD, 1,80,112, 4,3,1,1, 80,112, 0);
    ew_tanh_kernel<<<g4,256,0,stream>>>(tD, 4*S2);
    ew_gru_kernel<<<g4,256,0,stream>>>(tB, tD, h2, 4*S2);
    // GRU3: x=h2(4), h=h3(2)
    conv_launch(stream, h2,4, h3,2, P(22), P(19), tA, 1,80,112, 2,3,1,1, 80,112, 0);
    ew_sigmoid_kernel<<<g2,256,0,stream>>>(tA, 2*S2);
    conv_launch(stream, h2,4, h3,2, P(23), P(20), tB, 1,80,112, 2,3,1,1, 80,112, 0);
    ew_sigmoid_kernel<<<g2,256,0,stream>>>(tB, 2*S2);
    ew_mul_kernel<<<g2,256,0,stream>>>(tA, h3, tC, 2*S2);
    conv_launch(stream, h2,4, tC,2, P(21), P(18), tD, 1,80,112, 2,3,1,1, 80,112, 0);
    ew_tanh_kernel<<<g2,256,0,stream>>>(tD, 2*S2);
    ew_gru_kernel<<<g2,256,0,stream>>>(tB, tD, h3, 2*S2);
    // prob conv -> per-depth regularized score
    conv_launch(stream, h3,2, nullptr,0, P(25), P(24), regb + (size_t)d*S2,
                1,80,112, 1,3,1,1, 80,112, 0);
  }

  // ---------------- softmax over depth ----------------
  softmax_kernel<<<dim3((S2+255)/256), dim3(256), 0, stream>>>(
      regb, (float*)d_out, S2);
#undef CGR
}
